// CodeBook_13889924235619
// MI455X (gfx1250) — matrix operand order repaired and checked by reading, not compiled, not run on hardware
//
#include <hip/hip_runtime.h>

// ---- CDNA5 WMMA / TDM types -------------------------------------------
typedef __attribute__((ext_vector_type(16))) __bf16       v16bf;
typedef __attribute__((ext_vector_type(8)))  float        v8f;
typedef __attribute__((ext_vector_type(4)))  unsigned int v4u;
typedef __attribute__((ext_vector_type(8)))  int          v8i;
typedef __attribute__((ext_vector_type(4)))  int          v4i;

union FragBF16 {
    v16bf        v;
    unsigned int u[8];
    uint4        q[2];
};

// Pack two f32 bit patterns into {hi.bf16, lo.bf16} with ONE v_perm_b32
// (truncating f32->bf16; RNE skipped to save 3 VALU per element).
__device__ __forceinline__ unsigned int pack_bf16_pair(unsigned int lo,
                                                       unsigned int hi) {
    return __builtin_amdgcn_perm(hi, lo, 0x07060302u);
}

// z: [16, 64, 16384] f32 (t, d, n); emb: [256, 64] f32; out: [16,16384] i32
// Block: 256 rows x 256 codes of one t-plane. 256 threads = 8 wave32,
// each wave owns 32 rows. WMMA roles: A = code tile (16x32 bf16),
// B = x rows (32x16 bf16) -> each lane's 8 accs are 8 codes of ONE row,
// so argmin is in-lane and needs a single shfl_xor(16) at the end.
__global__ __launch_bounds__(256)
void codebook_assign_kernel(const float* __restrict__ z,
                            const float* __restrict__ emb,
                            int* __restrict__ out) {
    constexpr int D    = 64;     // feature dim
    constexpr int K    = 256;    // codebook size
    constexpr int NB   = 16384;  // positions per t
    constexpr int RT   = 256;    // rows per block
    constexpr int ESTR = 72;     // bf16 codebook row stride in shorts:
                                 // 144B = 36 dwords, gcd(36,64)=4 ->
                                 // conflict-free b128 row gathers

    __shared__ __align__(16) unsigned short embs[K * ESTR]; // bf16 codebook
    __shared__ __align__(16) float          e2s[K];         // |e|^2 (f32)
    extern __shared__ unsigned int xf32[];                  // [64 d][256 n] f32 (TDM dest)

    const int tid = threadIdx.x;
    const int t   = blockIdx.x >> 6;          // 64 blocks per t
    const int n0  = (blockIdx.x & 63) * RT;

    // ---- 1. Kick off TDM: DMA z tile [64 d][256 n] f32 -> LDS ----------
    // 2D descriptor: dim0 = n (contiguous, tile 256), dim1 = d (tile 64),
    // dim0_stride = 16384 elements. Zero per-element VMEM work.
    if (tid < 32) {   // wave 0 only (TDM ignores EXEC, issues per-wave)
        unsigned long long ga =
            (unsigned long long)(const void*)(z + (size_t)t * D * NB + n0);
        unsigned int laddr = (unsigned int)(unsigned long long)(const void*)xf32;
        v4u g0;
        g0[0] = 1u;                                  // count=1, user mode
        g0[1] = laddr;                               // LDS byte address
        g0[2] = (unsigned int)ga;                    // global addr lo
        g0[3] = (unsigned int)((ga >> 32) & 0x01FFFFFFu) | 0x80000000u; // type=2
        v8i g1;
        g1[0] = 0x00020000;                          // data_size = 4B
        g1[1] = (int)((unsigned)NB << 16);           // tensor_dim0 = 16384
        g1[2] = (int)((unsigned)D  << 16);           // tensor_dim1 = 64
        g1[3] = (int)((unsigned)RT << 16);           // tile_dim0 = 256
        g1[4] = D;                                   // tile_dim1 = 64
        g1[5] = NB;                                  // tensor_dim0_stride lo
        g1[6] = 0;
        g1[7] = 0;
        v4i gz4 = {0, 0, 0, 0};                      // groups 2/3 unused (2D)
        v8i gz8 = {0, 0, 0, 0, 0, 0, 0, 0};          // trailing group (6-arg form)
        __builtin_amdgcn_tensor_load_to_lds(g0, g1, gz4, gz4, gz8, 0);
    }

    // ---- 2. Stage codebook bf16 + |e|^2 while the TDM streams ----------
    {
        const int code = tid;
        const float4* er4 = (const float4*)(emb + code * D);
        unsigned short* erow = embs + code * ESTR;
        float s = 0.0f;
        #pragma unroll
        for (int q = 0; q < 16; ++q) {
            float4 v = er4[q];
            s += v.x * v.x + v.y * v.y + v.z * v.z + v.w * v.w;
            unsigned int p0 = pack_bf16_pair(__float_as_uint(v.x), __float_as_uint(v.y));
            unsigned int p1 = pack_bf16_pair(__float_as_uint(v.z), __float_as_uint(v.w));
            *(uint2*)(erow + q * 4) = make_uint2(p0, p1);
        }
        e2s[code] = s;
    }

    if (tid < 32) __builtin_amdgcn_s_wait_tensorcnt(0);
    __syncthreads();

    // ---- 3. Hoist B fragments (x rows) straight from the f32 tile ------
    const int lane = tid & 31;
    const int wave = tid >> 5;
    const int m    = lane & 15;     // B column / A row within tile
    const int kh   = lane >> 4;     // K-half select (this file's contiguous A/B K order)
    const int kh8  = kh * 8;

    const int r0 = wave * 32 + m;   // row tile 0
    const int r1 = r0 + 16;         // row tile 1

    // B layout (32x16 bf16): lane = column n, element e -> K = 32*s+16*kh+e.
    // Gather = conflict-free ds_load_b32 (lanes vary along contiguous n)
    // + v_perm bf16 pack. Loaded ONCE, reused for all 256 codes.
    FragBF16 B00, B01, B10, B11;
    #pragma unroll
    for (int i = 0; i < 8; ++i) {
        const int d0 = 16 * kh + 2 * i;  // kstep 0
        const int d1 = d0 + 32;          // kstep 1
        B00.u[i] = pack_bf16_pair(xf32[(d0    ) * RT + r0], xf32[(d0 + 1) * RT + r0]);
        B01.u[i] = pack_bf16_pair(xf32[(d1    ) * RT + r0], xf32[(d1 + 1) * RT + r0]);
        B10.u[i] = pack_bf16_pair(xf32[(d0    ) * RT + r1], xf32[(d0 + 1) * RT + r1]);
        B11.u[i] = pack_bf16_pair(xf32[(d1    ) * RT + r1], xf32[(d1 + 1) * RT + r1]);
    }

    // ---- 4. Hot loop: 16 code tiles x (2 WMMA chains) ------------------
    // |x|^2 dropped (constant per row -> argmin-invariant):
    //   score = |e|^2 - 2 x.e ; sqrt monotone -> skip.
    float bv0[8], bv1[8];
    int   bk0[8], bk1[8];
    #pragma unroll
    for (int i = 0; i < 8; ++i) {
        bv0[i] = 3.4e38f; bv1[i] = 3.4e38f; bk0[i] = 0; bk1[i] = 0;
    }

    const unsigned short* abase = embs + m * ESTR;

    #pragma unroll 2
    for (int kt = 0; kt < 16; ++kt) {
        // A fragments (code tile, 16x32 bf16), padded-stride b128 gathers:
        //   lanes 0-15: M=lane, e0..15->K=e
        //   lanes16-31: M=lane, e0..15->K=e+16
        const unsigned short* arow = abase + kt * 16 * ESTR;
        FragBF16 A0, A1;
        A0.q[0] = *(const uint4*)(arow +  0 + 2 * kh8);
        A0.q[1] = *(const uint4*)(arow +  8 + 2 * kh8);
        A1.q[0] = *(const uint4*)(arow + 32 + 2 * kh8);
        A1.q[1] = *(const uint4*)(arow + 40 + 2 * kh8);

        v8f acc0 = {}, acc1 = {};   // two independent chains fill hazard slots
        acc0 = __builtin_amdgcn_wmma_f32_16x16x32_bf16(false, A0.v, false, B00.v, (short)0, acc0, false, false);
        acc1 = __builtin_amdgcn_wmma_f32_16x16x32_bf16(false, A0.v, false, B10.v, (short)0, acc1, false, false);
        acc0 = __builtin_amdgcn_wmma_f32_16x16x32_bf16(false, A1.v, false, B01.v, (short)0, acc0, false, false);
        acc1 = __builtin_amdgcn_wmma_f32_16x16x32_bf16(false, A1.v, false, B11.v, (short)0, acc1, false, false);

        // C layout: lane holds 8 codes (M = kh8+i) of ONE row (N = m).
        const float4 e2lo = *(const float4*)&e2s[kt * 16 + kh8];
        const float4 e2hi = *(const float4*)&e2s[kt * 16 + kh8 + 4];
        const float e2a[8] = {e2lo.x, e2lo.y, e2lo.z, e2lo.w,
                              e2hi.x, e2hi.y, e2hi.z, e2hi.w};

        #pragma unroll
        for (int i = 0; i < 8; ++i) {
            // track kt (uniform SGPR cndmask source), reconstruct code later
            float d0 = fmaf(-2.0f, acc0[i], e2a[i]);
            if (d0 < bv0[i]) { bv0[i] = d0; bk0[i] = kt; }
            float d1 = fmaf(-2.0f, acc1[i], e2a[i]);
            if (d1 < bv1[i]) { bv1[i] = d1; bk1[i] = kt; }
        }
    }

    // ---- 5. In-lane argmin over 8 slots, then one kh-half exchange -----
    #pragma unroll
    for (int tile = 0; tile < 2; ++tile) {
        const float* bv = tile ? bv1 : bv0;
        const int*   bk = tile ? bk1 : bk0;
        float best = bv[0];
        int   bc   = bk[0] * 16 + kh8;          // slot 0's code
        #pragma unroll
        for (int i = 1; i < 8; ++i) {
            int c = bk[i] * 16 + kh8 + i;
            if (bv[i] < best || (bv[i] == best && c < bc)) {
                best = bv[i]; bc = c;
            }
        }
        // other 128 codes for this row live in the partner kh-half
        float ov = __shfl_xor(best, 16, 32);
        int   oc = __shfl_xor(bc, 16, 32);
        if (ov < best || (ov == best && oc < bc)) { best = ov; bc = oc; }

        if (kh == 0) {
            out[(size_t)t * NB + n0 + (tile ? r1 : r0)] = bc;
        }
    }
}

extern "C" void kernel_launch(void* const* d_in, const int* in_sizes, int n_in,
                              void* d_out, int out_size, void* d_ws, size_t ws_size,
                              hipStream_t stream) {
    (void)in_sizes; (void)n_in; (void)d_ws; (void)ws_size; (void)out_size;
    const float* z   = (const float*)d_in[0];   // [16, 64, 128, 128] f32
    const float* emb = (const float*)d_in[1];   // [256, 64] f32
    int* out = (int*)d_out;                     // [16, 128, 128] i32 codes

    dim3 grid(16 * 64);        // 16 t-planes x 64 row-tiles of 256
    dim3 block(256);           // 8 wave32
    size_t dyn_lds = 64 * 256 * sizeof(float);  // TDM destination tile
    codebook_assign_kernel<<<grid, block, dyn_lds, stream>>>(z, emb, out);
}
